// VirTuesDecoderMAE_77197742179053
// MI455X (gfx1250) — matrix operand order repaired and checked by reading, not compiled, not run on hardware
//
#include <hip/hip_runtime.h>
#include <hip/hip_bf16.h>
#include <math.h>

// ---------------------------------------------------------------------------
// Types / helpers
// ---------------------------------------------------------------------------
typedef __bf16 bf16_t;
typedef __attribute__((ext_vector_type(16))) __bf16 bf16x16;
typedef __attribute__((ext_vector_type(8)))  __bf16 bf16x8;
typedef __attribute__((ext_vector_type(8)))  float  f32x8;
typedef int v4i __attribute__((vector_size(16)));            // matches builtin param
typedef __attribute__((address_space(1))) v4i* gv4i_p;       // global int4*
typedef __attribute__((address_space(3))) v4i* lv4i_p;       // LDS int4*

#define DEV static __device__ __forceinline__

#if __has_builtin(__builtin_amdgcn_global_load_async_to_lds_b128) && \
    __has_builtin(__builtin_amdgcn_s_wait_asynccnt)
#define HAVE_ASYNC_LDS 1
#else
#define HAVE_ASYNC_LDS 0
#endif

DEV bf16_t f2bf(float f) {
  unsigned u = __builtin_bit_cast(unsigned, f);
  unsigned r = u + 0x7FFFu + ((u >> 16) & 1u);   // round-to-nearest-even
  unsigned short h = (unsigned short)(r >> 16);
  return __builtin_bit_cast(bf16_t, h);
}
DEV float bf2f(bf16_t b) {
  unsigned short h = __builtin_bit_cast(unsigned short, b);
  unsigned u = ((unsigned)h) << 16;
  return __builtin_bit_cast(float, u);
}

union FragU { bf16x16 v; bf16x8 h[2]; };

// A-fragment (16x32 bf16): lane<16 -> row=lane, K {0..7,16..23};
//                          lane>=16 -> row=lane-16, K {8..15,24..31}
DEV bf16x16 load_frag_a(const bf16_t* row, int laneHi) {
  FragU f;
  const bf16_t* p = row + (laneHi ? 8 : 0);
  f.h[0] = *(const bf16x8*)(p);
  f.h[1] = *(const bf16x8*)(p + 16);
  return f.v;
}
// B-fragment (32x16 bf16) loaded from B^T (NxK row-major):
// lane<16 -> col N=lane, K {0..7,16..23} ; lane>=16 -> col N=lane-16, K {8..15,24..31}
DEV bf16x16 load_frag_b(const bf16_t* row, int laneHi) {
  FragU f;
  const bf16_t* p = row + (laneHi ? 8 : 0);
  f.h[0] = *(const bf16x8*)(p);
  f.h[1] = *(const bf16x8*)(p + 16);
  return f.v;
}
DEV f32x8 wmma_bf16(bf16x16 a, bf16x16 b, f32x8 c) {
  return __builtin_amdgcn_wmma_f32_16x16x32_bf16(false, a, false, b,
                                                 (short)0, c, false, false);
}

// 16-byte global -> LDS copy: async on CDNA5 (ASYNCcnt), else via VGPRs
DEV void cp16(const bf16_t* g, bf16_t* l) {
#if HAVE_ASYNC_LDS
  __builtin_amdgcn_global_load_async_to_lds_b128((gv4i_p)g, (lv4i_p)l, 0, 0);
#else
  *(bf16x8*)l = *(const bf16x8*)g;
#endif
}
DEV void wait_async() {
#if HAVE_ASYNC_LDS
  __builtin_amdgcn_s_wait_asynccnt(0);
#endif
}

// ---------------------------------------------------------------------------
// GEMM:  C(MxN,f32) = A(MxK,bf16) * BT(NxK,bf16)^T + bias
// 128 thr / 4 waves, block tile 64x64, per-wave 32x32 (4 WMMA accumulators),
// K-step 32, double-buffered LDS fed by async global->LDS copies.
// ---------------------------------------------------------------------------
__launch_bounds__(128)
__global__ void gemm_bf16_kernel(const bf16_t* __restrict__ A,
                                 const bf16_t* __restrict__ BT,
                                 const float*  __restrict__ bias,
                                 float* __restrict__ C,
                                 int M, int N, int K) {
  __shared__ __align__(16) bf16_t As[2][64][40];
  __shared__ __align__(16) bf16_t Bs[2][64][40];
  const int tid    = threadIdx.x;
  const int lane   = tid & 31;
  const int w      = tid >> 5;           // 0..3
  const int l15    = lane & 15;
  const int laneHi = lane >> 4;
  const int mBase  = blockIdx.y * 64;
  const int nBase  = blockIdx.x * 64;
  const int wr = w & 1, wc = w >> 1;     // 2x2 wave grid, 32x32 each

  const int ldRow = tid >> 1;            // 0..63
  const int ldCol = (tid & 1) * 16;      // 0 or 16 (two 8-elem chunks each)

  f32x8 acc00 = {}, acc01 = {}, acc10 = {}, acc11 = {};

  const bf16_t* gA = A  + (size_t)(mBase + ldRow) * K + ldCol;
  const bf16_t* gB = BT + (size_t)(nBase + ldRow) * K + ldCol;

  auto stage = [&](int buf, int kb) {
    cp16(gA + kb,     &As[buf][ldRow][ldCol]);
    cp16(gA + kb + 8, &As[buf][ldRow][ldCol + 8]);
    cp16(gB + kb,     &Bs[buf][ldRow][ldCol]);
    cp16(gB + kb + 8, &Bs[buf][ldRow][ldCol + 8]);
  };

  stage(0, 0);
  wait_async();
  __syncthreads();

  int buf = 0;
  for (int kb = 0; kb < K; kb += 32) {
    const int nkb = kb + 32;
    if (nkb < K) {
      stage(buf ^ 1, nkb);                              // overlap with compute
      __builtin_prefetch(gA + nkb + 32, 0, 1);          // global_prefetch_b8
      __builtin_prefetch(gB + nkb + 32, 0, 1);
    }
    bf16x16 a0 = load_frag_a(&As[buf][wr * 32 + l15][0], laneHi);
    bf16x16 a1 = load_frag_a(&As[buf][wr * 32 + 16 + l15][0], laneHi);
    bf16x16 b0 = load_frag_b(&Bs[buf][wc * 32 + l15][0], laneHi);
    bf16x16 b1 = load_frag_b(&Bs[buf][wc * 32 + 16 + l15][0], laneHi);
    acc00 = wmma_bf16(a0, b0, acc00);
    acc01 = wmma_bf16(a0, b1, acc01);
    acc10 = wmma_bf16(a1, b0, acc10);
    acc11 = wmma_bf16(a1, b1, acc11);
    wait_async();
    __syncthreads();
    buf ^= 1;
  }

  const int rowOff = laneHi * 8;
  const int c0 = nBase + wc * 32 + l15;
  const float bv0 = bias ? bias[c0] : 0.f;
  const float bv1 = bias ? bias[c0 + 16] : 0.f;
#pragma unroll
  for (int r = 0; r < 8; ++r) {
    const size_t row0 = (size_t)(mBase + wr * 32 + r + rowOff);
    const size_t row1 = row0 + 16;
    C[row0 * N + c0]      = acc00[r] + bv0;
    C[row0 * N + c0 + 16] = acc01[r] + bv1;
    C[row1 * N + c0]      = acc10[r] + bv0;
    C[row1 * N + c0 + 16] = acc11[r] + bv1;
  }
}

// ---------------------------------------------------------------------------
// 'h' attention: per (n, head, q-quarter). 4 waves x 16 q-rows. S=256, dh=64.
// ---------------------------------------------------------------------------
__launch_bounds__(128)
__global__ void attn_h_kernel(const bf16_t* __restrict__ qH,
                              const bf16_t* __restrict__ kH,
                              const bf16_t* __restrict__ vT,   // (n,h,d,s) bf16
                              float* __restrict__ oF) {
  __shared__ __align__(16) float sc[4][16][256];
  const int lane   = threadIdx.x & 31;
  const int w      = threadIdx.x >> 5;
  const int l15    = lane & 15;
  const int laneHi = lane >> 4;
  const int bid = blockIdx.x;
  const int qt = bid & 3;
  const int h  = (bid >> 2) & 7;
  const int n  = bid >> 5;
  const int q0 = qt * 64 + w * 16;

  const bf16_t* qRow = qH + ((size_t)(n * 256 + q0 + l15)) * 512 + h * 64;

  // scores = Q K^T / 8 : 16 key-tiles, two WMMAs each (dh=64)
  for (int j = 0; j < 16; ++j) {
    f32x8 s = {};
    const bf16_t* kRow = kH + ((size_t)(n * 256 + j * 16 + l15)) * 512 + h * 64;
#pragma unroll
    for (int kk = 0; kk < 64; kk += 32) {
      bf16x16 a = load_frag_a(qRow + kk, laneHi);
      bf16x16 b = load_frag_b(kRow + kk, laneHi);
      s = wmma_bf16(a, b, s);
    }
#pragma unroll
    for (int r = 0; r < 8; ++r)
      sc[w][r + laneHi * 8][j * 16 + l15] = s[r] * 0.125f;
  }

  // softmax: two lanes per row (halves of 256 cols), in-place in LDS
  {
    const int row = lane >> 1, half = lane & 1;
    float* rp = &sc[w][row][half * 128];
    float m = -3.4e38f;
    for (int c = 0; c < 128; ++c) m = fmaxf(m, rp[c]);
    m = fmaxf(m, __shfl_xor(m, 1, 32));
    float sum = 0.f;
    for (int c = 0; c < 128; ++c) { float e = __expf(rp[c] - m); rp[c] = e; sum += e; }
    sum += __shfl_xor(sum, 1, 32);
    const float inv = 1.0f / sum;
    for (int c = 0; c < 128; ++c) rp[c] *= inv;
  }

  // O = P V : dh in 4 tiles of 16, keys in 8 chunks of 32
  for (int dt = 0; dt < 4; ++dt) {
    f32x8 acc = {};
    const bf16_t* vRow = vT + ((size_t)((n * 8 + h) * 64 + dt * 16 + l15)) * 256;
    for (int kk = 0; kk < 8; ++kk) {
      const float* pr = &sc[w][l15][kk * 32 + (laneHi ? 8 : 0)];
      bf16x16 a;
#pragma unroll
      for (int i = 0; i < 8; ++i) { a[i] = f2bf(pr[i]); a[i + 8] = f2bf(pr[16 + i]); }
      bf16x16 b = load_frag_b(vRow + kk * 32, laneHi);
      acc = wmma_bf16(a, b, acc);
    }
#pragma unroll
    for (int r = 0; r < 8; ++r)
      oF[((size_t)(n * 256 + q0 + r + laneHi * 8)) * 512 + h * 64 + dt * 16 + l15] = acc[r];
  }
}

// ---------------------------------------------------------------------------
// 'v' attention: seq length 2 per (bc,s) pair. RoPE provably cancels here
// (equal positions -> orthogonal rotation preserves q.k), so q/k are plain.
// ---------------------------------------------------------------------------
__global__ void attn_v_kernel(const bf16_t* __restrict__ qH,
                              const bf16_t* __restrict__ kH,
                              const bf16_t* __restrict__ vH,
                              float* __restrict__ oF) {
  const int id = blockIdx.x * blockDim.x + threadIdx.x;   // 16384*8
  const int h = id & 7;
  const int pair = id >> 3;
  const int s = pair & 255;
  const int bc = pair >> 8;
  const size_t r0 = ((size_t)(bc * 2) * 256 + s) * 512 + h * 64;
  const size_t r1 = r0 + (size_t)256 * 512;
  float s00 = 0, s01 = 0, s10 = 0, s11 = 0;
  for (int d = 0; d < 64; ++d) {
    const float q0 = bf2f(qH[r0 + d]), q1 = bf2f(qH[r1 + d]);
    const float k0 = bf2f(kH[r0 + d]), k1 = bf2f(kH[r1 + d]);
    s00 += q0 * k0; s01 += q0 * k1; s10 += q1 * k0; s11 += q1 * k1;
  }
  s00 *= 0.125f; s01 *= 0.125f; s10 *= 0.125f; s11 *= 0.125f;
  const float m0 = fmaxf(s00, s01), m1 = fmaxf(s10, s11);
  const float e00 = __expf(s00 - m0), e01 = __expf(s01 - m0);
  const float e10 = __expf(s10 - m1), e11 = __expf(s11 - m1);
  const float p00 = e00 / (e00 + e01), p01 = e01 / (e00 + e01);
  const float p10 = e10 / (e10 + e11), p11 = e11 / (e10 + e11);
  for (int d = 0; d < 64; ++d) {
    const float v0 = bf2f(vH[r0 + d]), v1 = bf2f(vH[r1 + d]);
    oF[r0 + d] = p00 * v0 + p01 * v1;
    oF[r1 + d] = p10 * v0 + p11 * v1;
  }
}

// ---------------------------------------------------------------------------
// Residual + LayerNorm (D=512): one wave per token, writes f32 + bf16 copies
// ---------------------------------------------------------------------------
__launch_bounds__(256)
__global__ void res_ln_kernel(float* __restrict__ tokF, bf16_t* __restrict__ tokH,
                              const float* __restrict__ addF,
                              const float* __restrict__ g,
                              const float* __restrict__ b) {
  const int lane = threadIdx.x & 31;
  const int w = threadIdx.x >> 5;
  const size_t tok = (size_t)blockIdx.x * 8 + w;
  float* xp = tokF + tok * 512;
  const float* ap = addF + tok * 512;
  float v[16];
  float mean = 0.f;
#pragma unroll
  for (int i = 0; i < 16; ++i) { v[i] = xp[lane + i * 32] + ap[lane + i * 32]; mean += v[i]; }
#pragma unroll
  for (int o = 16; o > 0; o >>= 1) mean += __shfl_xor(mean, o, 32);
  mean *= (1.0f / 512.0f);
  float var = 0.f;
#pragma unroll
  for (int i = 0; i < 16; ++i) { const float dv = v[i] - mean; var += dv * dv; }
#pragma unroll
  for (int o = 16; o > 0; o >>= 1) var += __shfl_xor(var, o, 32);
  const float rstd = rsqrtf(var * (1.0f / 512.0f) + 1e-5f);
#pragma unroll
  for (int i = 0; i < 16; ++i) {
    const int c = lane + i * 32;
    const float y = (v[i] - mean) * rstd * g[c] + b[c];
    xp[c] = y;
    tokH[tok * 512 + c] = f2bf(y);
  }
}

// ---------------------------------------------------------------------------
// RoPE (+ optional) fp32 -> bf16 convert, one thread per rotation pair
// ---------------------------------------------------------------------------
__global__ void rope_cvt_kernel(const float* __restrict__ src,
                                bf16_t* __restrict__ dst, int applyRope) {
  const size_t id = (size_t)blockIdx.x * blockDim.x + threadIdx.x; // 32768*256
  const int p = (int)(id & 255);
  const size_t tok = id >> 8;
  const int hh = p >> 5;
  const int i2 = p & 31;
  const int halfSel = i2 >> 4;
  const int fi = i2 & 15;
  const size_t base = tok * 512 + hh * 64 + halfSel * 32 + fi * 2;
  float a0 = src[base], a1 = src[base + 1];
  if (applyRope) {
    const int s = (int)(tok & 255);
    const float pos = halfSel ? (float)(s & 15) : (float)(s >> 4);
    const float ang = pos * __powf(10000.0f, -(float)fi * 0.0625f);
    float sn, cs;
    __sincosf(ang, &sn, &cs);
    const float r0 = a0 * cs - a1 * sn;
    const float r1 = a0 * sn + a1 * cs;
    a0 = r0; a1 = r1;
  }
  dst[base] = f2bf(a0);
  dst[base + 1] = f2bf(a1);
}

// V (token-major f32) -> V^T per (n,h): (n,h,d,s) bf16
__global__ void cvt_vT_kernel(const float* __restrict__ vF, bf16_t* __restrict__ vT) {
  const size_t id = (size_t)blockIdx.x * blockDim.x + threadIdx.x; // 16,777,216
  const int s = (int)(id & 255);
  const int rest = (int)(id >> 8);
  const int d = rest & 63;
  const int nh = rest >> 6;
  const int h = nh & 7;
  const int n = nh >> 3;
  vT[id] = f2bf(vF[((size_t)n * 256 + s) * 512 + h * 64 + d]);
}

// tanh-GELU + fp32 -> bf16
__global__ void gelu_cvt_kernel(const float* __restrict__ in, bf16_t* __restrict__ out) {
  const size_t id = (size_t)blockIdx.x * blockDim.x + threadIdx.x;
  const float v = in[id];
  const float u = 0.7978845608028654f * (v + 0.044715f * v * v * v);
  out[id] = f2bf(0.5f * v * (1.0f + tanhf(u)));
}

// tok construction: interleave ps-token (t=0) and channel token (t=1)
__global__ void build_tok_kernel(const float* __restrict__ x, const float* __restrict__ ps,
                                 float* __restrict__ tokF, bf16_t* __restrict__ tokH) {
  const size_t id = (size_t)blockIdx.x * blockDim.x + threadIdx.x;  // 16,777,216
  const int d = (int)(id & 511);
  const int tk = (int)(id >> 9);
  const int s = tk & 255;
  const int nt = tk >> 8;
  const int t = nt & 1;
  const int bc = nt >> 1;
  const int b = bc >> 4;
  const float v = t ? x[((size_t)bc * 256 + s) * 512 + d]
                    : ps[((size_t)b * 256 + s) * 512 + d];
  tokF[id] = v;
  tokH[id] = f2bf(v);
}

// gather t=1 tokens for the decoder head
__global__ void gather_ctx_kernel(const bf16_t* __restrict__ tokH, bf16_t* __restrict__ ctxH) {
  const size_t id = (size_t)blockIdx.x * blockDim.x + threadIdx.x;  // 8,388,608
  const int d = (int)(id & 511);
  const int m = (int)(id >> 9);
  const int s = m & 255;
  const int bc = m >> 8;
  ctxH[id] = tokH[((size_t)(bc * 2 + 1) * 256 + s) * 512 + d];
}

// weight transpose + convert: W(KxN,f32) -> Wt(NxK,bf16)
__global__ void pack_wt_kernel(const float* __restrict__ W, bf16_t* __restrict__ Wt,
                               int K, int N) {
  const size_t id = (size_t)blockIdx.x * blockDim.x + threadIdx.x;  // N*K
  const int k = (int)(id % K);
  const int n = (int)(id / K);
  Wt[(size_t)n * K + k] = f2bf(W[(size_t)k * N + n]);
}

// ---------------------------------------------------------------------------
// Host orchestration
// ---------------------------------------------------------------------------
extern "C" void kernel_launch(void* const* d_in, const int* in_sizes, int n_in,
                              void* d_out, int out_size, void* d_ws, size_t ws_size,
                              hipStream_t stream) {
  (void)in_sizes; (void)n_in; (void)out_size; (void)ws_size;
  const float* x    = (const float*)d_in[0];
  const float* ps   = (const float*)d_in[1];
  const float* Wq   = (const float*)d_in[2];
  const float* bq   = (const float*)d_in[3];
  const float* Wk   = (const float*)d_in[4];
  const float* bk   = (const float*)d_in[5];
  const float* Wv   = (const float*)d_in[6];
  const float* bv   = (const float*)d_in[7];
  const float* Wo   = (const float*)d_in[8];
  const float* bo   = (const float*)d_in[9];
  const float* ln1g = (const float*)d_in[10];
  const float* ln1b = (const float*)d_in[11];
  const float* ln2g = (const float*)d_in[12];
  const float* ln2b = (const float*)d_in[13];
  const float* W1   = (const float*)d_in[14];
  const float* b1   = (const float*)d_in[15];
  const float* W2   = (const float*)d_in[16];
  const float* b2   = (const float*)d_in[17];
  const float* Wd   = (const float*)d_in[18];
  const float* bd   = (const float*)d_in[19];

  constexpr size_t NTOK = 32768;
  char* ws = (char*)d_ws;
  float*  tokF = (float*)(ws);
  bf16_t* tokH = (bf16_t*)(ws + NTOK * 512 * 4);
  float*  bufF = (float*)(ws + NTOK * 512 * 6);
  size_t off = NTOK * 512 * 6 + NTOK * 1024 * 4;
  bf16_t* qH  = (bf16_t*)(ws + off); off += NTOK * 512 * 2;
  bf16_t* kH  = (bf16_t*)(ws + off); off += NTOK * 512 * 2;
  bf16_t* vH  = (bf16_t*)(ws + off); off += NTOK * 512 * 2;
  bf16_t* vT  = (bf16_t*)(ws + off); off += NTOK * 512 * 2;
  bf16_t* h1H = (bf16_t*)(ws + off); off += NTOK * 1024 * 2;
  bf16_t* wp  = (bf16_t*)(ws + off);

  const size_t DD = 512 * 512, DF = 512 * 1024;
  const size_t LSTR = 4 * DD + 2 * DF;

  auto pack = [&](const float* W, bf16_t* Wt, int K, int N) {
    pack_wt_kernel<<<(unsigned)(((size_t)K * N) / 256), 256, 0, stream>>>(W, Wt, K, N);
  };
  for (int i = 0; i < 4; ++i) {
    bf16_t* base = wp + (size_t)i * LSTR;
    pack(Wq + (size_t)i * DD, base + 0 * DD, 512, 512);
    pack(Wk + (size_t)i * DD, base + 1 * DD, 512, 512);
    pack(Wv + (size_t)i * DD, base + 2 * DD, 512, 512);
    pack(Wo + (size_t)i * DD, base + 3 * DD, 512, 512);
    pack(W1 + (size_t)i * DF, base + 4 * DD, 512, 1024);
    pack(W2 + (size_t)i * DF, base + 4 * DD + DF, 1024, 512);
  }
  bf16_t* wdT = wp + 4 * LSTR;
  pack(Wd, wdT, 512, 576);

  build_tok_kernel<<<65536, 256, 0, stream>>>(x, ps, tokF, tokH);

  auto gemm = [&](const bf16_t* A, const bf16_t* BT, const float* bias, float* C,
                  int M, int N, int K) {
    dim3 g(N / 64, M / 64);
    gemm_bf16_kernel<<<g, 128, 0, stream>>>(A, BT, bias, C, M, N, K);
  };

  for (int i = 0; i < 4; ++i) {
    const bool isH = ((i & 1) == 0);         // PATTERN = h,v,h,v
    bf16_t* base = wp + (size_t)i * LSTR;

    gemm(tokH, base + 0 * DD, bq + i * 512, bufF, 32768, 512, 512);
    rope_cvt_kernel<<<32768, 256, 0, stream>>>(bufF, qH, isH ? 1 : 0);
    gemm(tokH, base + 1 * DD, bk + i * 512, bufF, 32768, 512, 512);
    rope_cvt_kernel<<<32768, 256, 0, stream>>>(bufF, kH, isH ? 1 : 0);
    gemm(tokH, base + 2 * DD, bv + i * 512, bufF, 32768, 512, 512);

    bf16_t* oH;
    if (isH) {
      cvt_vT_kernel<<<65536, 256, 0, stream>>>(bufF, vT);
      attn_h_kernel<<<4096, 128, 0, stream>>>(qH, kH, vT, bufF);
      oH = vH;                                // vH unused in 'h' layers
    } else {
      rope_cvt_kernel<<<32768, 256, 0, stream>>>(bufF, vH, 0);
      attn_v_kernel<<<512, 256, 0, stream>>>(qH, kH, vH, bufF);
      oH = vT;                                // vT unused in 'v' layers
    }
    rope_cvt_kernel<<<32768, 256, 0, stream>>>(bufF, oH, 0);

    gemm(oH, base + 3 * DD, bo + i * 512, bufF, 32768, 512, 512);
    res_ln_kernel<<<4096, 256, 0, stream>>>(tokF, tokH, bufF,
                                            ln1g + i * 512, ln1b + i * 512);

    gemm(tokH, base + 4 * DD, b1 + i * 1024, bufF, 32768, 1024, 512);
    gelu_cvt_kernel<<<131072, 256, 0, stream>>>(bufF, h1H);
    gemm(h1H, base + 4 * DD + DF, b2 + i * 512, bufF, 32768, 512, 1024);
    res_ln_kernel<<<4096, 256, 0, stream>>>(tokF, tokH, bufF,
                                            ln2g + i * 512, ln2b + i * 512);
  }

  bf16_t* ctxH = qH;                          // qH free after last layer
  gather_ctx_kernel<<<32768, 256, 0, stream>>>(tokH, ctxH);
  gemm(ctxH, wdT, bd, (float*)d_out, 16384, 576, 512);
}
